// SPDetector_32693291057207
// MI455X (gfx1250) — matrix operand order repaired and checked by reading, not compiled, not run on hardware
//
#include <hip/hip_runtime.h>
#include <hip/hip_bf16.h>
#include <math.h>

// ---------------------------------------------------------------------------
// CDNA5 (gfx1250) SuperPoint-style detector: bf16 WMMA implicit-GEMM convs +
// fused softmax/pixel-shuffle + separable-LDS grid NMS.
// wave32 everywhere; WMMA fragment layouts per CDNA5 ISA 7.12.2.
// ---------------------------------------------------------------------------

typedef __attribute__((ext_vector_type(16))) __bf16 v16bf;
typedef __attribute__((ext_vector_type(8)))  __bf16 v8bf;
typedef __attribute__((ext_vector_type(8)))  float  v8f;
typedef __attribute__((ext_vector_type(4)))  float  v4f;

#define B_    32
#define C_    128
#define H_    64
#define W_    64
#define MID_  256
#define NCLS_ 65
#define MPAD_ 80
#define HI_   512
#define NPIX  (B_ * HI_ * HI_)      // 8388608 output pixels
#define NPOS  (B_ * H_ * W_)        // 131072 coarse positions

__device__ __forceinline__ v8f wmma_bf16(v16bf a, v16bf b, v8f c) {
  // v_wmma_f32_16x16x32_bf16 : D = A(16x32) * B(32x16) + C(16x16 f32)
  return __builtin_amdgcn_wmma_f32_16x16x32_bf16(
      /*neg_a=*/false, a, /*neg_b=*/false, b,
      /*c_mod=*/(short)0, c, /*reuse_a=*/false, /*reuse_b=*/false);
}

// Build a 16-element bf16 fragment from two contiguous 16B chunks.
__device__ __forceinline__ v16bf ldpair(const __bf16* p0, const __bf16* p1) {
  v8bf a = *(const v8bf*)p0;
  v8bf b = *(const v8bf*)p1;
  v16bf r;
#pragma unroll
  for (int i = 0; i < 8; ++i) { r[i] = a[i]; r[i + 8] = b[i]; }
  return r;
}

// ---------------------------------------------------------------------------
// Weight / bias prep: f32 -> bf16, tap-major W1, zero-padded W2, -inf bias pad
// ---------------------------------------------------------------------------
__global__ void prep_w1_kernel(const float* __restrict__ w1, __bf16* __restrict__ W1b) {
  int idx = blockIdx.x * 256 + threadIdx.x;          // 9*256*128 = 294912
  if (idx >= 9 * MID_ * C_) return;
  int tap = idx / (MID_ * C_);
  int rem = idx % (MID_ * C_);
  int o = rem / C_, c = rem % C_;
  int dy = tap / 3, dx = tap % 3;
  W1b[idx] = (__bf16)w1[((o * C_ + c) * 3 + dy) * 3 + dx];
}

__global__ void prep_w2_kernel(const float* __restrict__ w2, __bf16* __restrict__ W2b) {
  int idx = blockIdx.x * 256 + threadIdx.x;          // 80*256 = 20480
  if (idx >= MPAD_ * MID_) return;
  int o = idx / MID_, c = idx % MID_;
  W2b[idx] = (o < NCLS_) ? (__bf16)w2[o * MID_ + c] : (__bf16)0.0f;
}

__global__ void prep_b2_kernel(const float* __restrict__ b2, float* __restrict__ b2p) {
  int o = threadIdx.x;
  if (o < MPAD_) b2p[o] = (o < NCLS_) ? b2[o] : -1.0e30f;
}

// ---------------------------------------------------------------------------
// conv1: 3x3 conv (128->256) + bias + ReLU, bf16 WMMA implicit GEMM.
// One workgroup (8 waves) per (b, y) image row: 64 positions x 256 ochans.
// Wave w: 32 ochans x 64 positions -> 8 accumulator tiles.
// ---------------------------------------------------------------------------
__global__ __launch_bounds__(256) void conv1_kernel(
    const float* __restrict__ x, const __bf16* __restrict__ W1b,
    const float* __restrict__ b1, __bf16* __restrict__ Hb) {
  // channel dim padded 128->136 to stagger LDS banks (272B row stride)
  __shared__ __bf16 xs[3][66][136];

  const int tid  = threadIdx.x;
  const int lane = tid & 31;
  const int wv   = tid >> 5;
  const int b    = blockIdx.x >> 6;
  const int y    = blockIdx.x & 63;

  // Stage 3-row halo slab (cols -1..64, zero padded) as bf16.
  for (int e = tid; e < 3 * 66 * C_; e += 256) {
    int j = e % 66;
    int c = (e / 66) % C_;
    int r = e / (66 * C_);
    int gy = y + r - 1, gx = j - 1;
    float v = 0.0f;
    if (gy >= 0 && gy < H_ && gx >= 0 && gx < W_)
      v = x[(((size_t)b * C_ + c) * H_ + gy) * W_ + gx];
    xs[r][j][c] = (__bf16)v;
  }
  __syncthreads();

  const int m16  = lane & 15;
  const int hi   = lane >> 4;
  const int hi8  = hi * 8;
  const int hi16 = hi * 16;
  const int obase = wv * 32;

  const v8f vz = {0.f, 0.f, 0.f, 0.f, 0.f, 0.f, 0.f, 0.f};
  v8f acc[2][4];
#pragma unroll
  for (int mt = 0; mt < 2; ++mt)
#pragma unroll
    for (int nt = 0; nt < 4; ++nt) acc[mt][nt] = vz;

  for (int tap = 0; tap < 9; ++tap) {
    const int dy  = tap / 3;
    const int dxo = tap % 3;
    const __bf16* Wp = W1b + (size_t)tap * MID_ * C_;   // [o][c] row-major
    for (int kc = 0; kc < 4; ++kc) {
      const int c0 = kc * 32;
      const __bf16* a0p = Wp + (size_t)(obase + m16) * C_ + c0 + hi8;
      const __bf16* a1p = Wp + (size_t)(obase + 16 + m16) * C_ + c0 + hi8;
      v16bf a0 = ldpair(a0p, a0p + 16);
      v16bf a1 = ldpair(a1p, a1p + 16);
#pragma unroll
      for (int nt = 0; nt < 4; ++nt) {
        const int j = nt * 16 + m16 + dxo;              // staged col for this N lane
        const __bf16* bp = &xs[dy][j][c0 + hi8];
        v16bf bb = ldpair(bp, bp + 16);
        acc[0][nt] = wmma_bf16(a0, bb, acc[0][nt]);
        acc[1][nt] = wmma_bf16(a1, bb, acc[1][nt]);
      }
    }
  }

  // Epilogue: bias + ReLU -> bf16 NHWC [b][y][x][o]; 8 consecutive o per lane.
#pragma unroll
  for (int mt = 0; mt < 2; ++mt) {
    const int ob = obase + mt * 16 + hi8;
    const float* bp = b1 + ob;
#pragma unroll
    for (int nt = 0; nt < 4; ++nt) {
      const int p = nt * 16 + m16;
      v8bf o8;
#pragma unroll
      for (int r = 0; r < 8; ++r) {
        float v = acc[mt][nt][r] + bp[r];
        v = v > 0.f ? v : 0.f;
        o8[r] = (__bf16)v;
      }
      *(v8bf*)(Hb + ((size_t)blockIdx.x * W_ + p) * MID_ + ob) = o8;
    }
  }
}

// ---------------------------------------------------------------------------
// conv2: 1x1 conv (256->65, padded to 80) + softmax(65) + dust drop +
// threshold + 8x8 pixel shuffle, written straight into the 512x512 score map.
// Each wave: 16 positions x 80 (padded) classes = 5 WMMA tiles.
// ---------------------------------------------------------------------------
__global__ __launch_bounds__(256) void conv2_kernel(
    const __bf16* __restrict__ Hb, const __bf16* __restrict__ W2b,
    const float* __restrict__ b2p, float* __restrict__ scores) {
  const int tid  = threadIdx.x;
  const int lane = tid & 31;
  const int wv   = tid >> 5;
  const int m16  = lane & 15;
  const int hi   = lane >> 4;
  const int hi8  = hi * 8;
  const int hi16 = hi * 16;
  const int pos  = blockIdx.x * 128 + wv * 16 + m16;   // this lane's position

  const v8f vz = {0.f, 0.f, 0.f, 0.f, 0.f, 0.f, 0.f, 0.f};
  v8f acc[5];
#pragma unroll
  for (int m = 0; m < 5; ++m) acc[m] = vz;

  for (int kc = 0; kc < 8; ++kc) {
    const int c0 = kc * 32;
    const __bf16* bp = Hb + (size_t)pos * MID_ + c0 + hi8;
    v16bf bb = ldpair(bp, bp + 16);
#pragma unroll
    for (int m = 0; m < 5; ++m) {
      const __bf16* ap = W2b + (size_t)(m * 16 + m16) * MID_ + c0 + hi8;
      v16bf aa = ldpair(ap, ap + 16);
      acc[m] = wmma_bf16(aa, bb, acc[m]);
    }
  }

  // 80 logits for this position live in lane pair (p, p+16): softmax via xor-16.
  float val[40];
  float vmax = -INFINITY;
#pragma unroll
  for (int m = 0; m < 5; ++m)
#pragma unroll
    for (int r = 0; r < 8; ++r) {
      int o = m * 16 + hi8 + r;
      float v = acc[m][r] + b2p[o];    // pad rows get -1e30 -> exp = 0
      val[m * 8 + r] = v;
      vmax = fmaxf(vmax, v);
    }
  vmax = fmaxf(vmax, __shfl_xor(vmax, 16));
  float s = 0.f;
#pragma unroll
  for (int i = 0; i < 40; ++i) {
    float e = __expf(val[i] - vmax);
    val[i] = e;
    s += e;
  }
  s += __shfl_xor(s, 16);
  const float inv = 1.0f / s;

  const int bb_ = pos >> 12;
  const int yy  = (pos >> 6) & 63;
  const int xx  = pos & 63;
  // m=4 holds o>=64 (dust + pad) -> dropped. o = 16m + 8*hi + r, dy = 2m+hi, dx = r.
#pragma unroll
  for (int m = 0; m < 4; ++m) {
    const int Y = yy * 8 + 2 * m + hi;
    float* op = scores + ((size_t)bb_ * HI_ + Y) * HI_ + (size_t)xx * 8;
    v4f lo, hiv;
#pragma unroll
    for (int r = 0; r < 4; ++r) {
      float h = val[m * 8 + r] * inv;
      lo[r] = (h >= 0.015f) ? h : 0.f;
    }
#pragma unroll
    for (int r = 4; r < 8; ++r) {
      float h = val[m * 8 + r] * inv;
      hiv[r - 4] = (h >= 0.015f) ? h : 0.f;
    }
    *(v4f*)op = lo;
    *(v4f*)(op + 4) = hiv;
  }
}

// ---------------------------------------------------------------------------
// Grid NMS: separable 9x9 max-pool stencils, 32x32 tiles + halo 4 in LDS.
// ---------------------------------------------------------------------------
#define NTB 32
#define HALO 4
#define TS (NTB + 2 * HALO)   // 40

// mask op: m = (src == maxpool9(src)) & (src > 0); orMode -> mask |= m
__global__ __launch_bounds__(256) void nms_max_kernel(
    const float* __restrict__ src, unsigned char* __restrict__ mask, int orMode) {
  __shared__ float xs[TS][TS];
  __shared__ float rm[TS][NTB];
  const int b = blockIdx.z;
  const int ty0 = blockIdx.y * NTB, tx0 = blockIdx.x * NTB;
  const float* S = src + (size_t)b * HI_ * HI_;

  for (int e = threadIdx.x; e < TS * TS; e += 256) {
    int r = e / TS, c = e % TS;
    int gy = ty0 + r - HALO, gx = tx0 + c - HALO;
    float v = -INFINITY;
    if (gy >= 0 && gy < HI_ && gx >= 0 && gx < HI_) v = S[(size_t)gy * HI_ + gx];
    xs[r][c] = v;
  }
  __syncthreads();
  for (int e = threadIdx.x; e < TS * NTB; e += 256) {
    int r = e / NTB, c = e % NTB;
    float m = xs[r][c];
#pragma unroll
    for (int k = 1; k < 9; ++k) m = fmaxf(m, xs[r][c + k]);
    rm[r][c] = m;
  }
  __syncthreads();
  unsigned char* M = mask + (size_t)b * HI_ * HI_;
  for (int e = threadIdx.x; e < NTB * NTB; e += 256) {
    int r = e / NTB, c = e % NTB;
    float m = rm[r][c];
#pragma unroll
    for (int k = 1; k < 9; ++k) m = fmaxf(m, rm[r + k][c]);
    float v = xs[r + HALO][c + HALO];
    unsigned char nm = (v == m && v > 0.f) ? 1 : 0;
    size_t gi = (size_t)(ty0 + r) * HI_ + (tx0 + c);
    M[gi] = orMode ? (unsigned char)(M[gi] | nm) : nm;
  }
}

// t = dilate9(mask) ? 0 : scores
__global__ __launch_bounds__(256) void suppress_kernel(
    const unsigned char* __restrict__ mask, const float* __restrict__ scores,
    float* __restrict__ t) {
  __shared__ unsigned char ms[TS][TS];
  __shared__ unsigned char rm[TS][NTB];
  const int b = blockIdx.z;
  const int ty0 = blockIdx.y * NTB, tx0 = blockIdx.x * NTB;
  const unsigned char* M = mask + (size_t)b * HI_ * HI_;

  for (int e = threadIdx.x; e < TS * TS; e += 256) {
    int r = e / TS, c = e % TS;
    int gy = ty0 + r - HALO, gx = tx0 + c - HALO;
    unsigned char v = 0;
    if (gy >= 0 && gy < HI_ && gx >= 0 && gx < HI_) v = M[(size_t)gy * HI_ + gx];
    ms[r][c] = v;
  }
  __syncthreads();
  for (int e = threadIdx.x; e < TS * NTB; e += 256) {
    int r = e / NTB, c = e % NTB;
    unsigned char m = ms[r][c];
#pragma unroll
    for (int k = 1; k < 9; ++k) m |= ms[r][c + k];
    rm[r][c] = m;
  }
  __syncthreads();
  const float* S = scores + (size_t)b * HI_ * HI_;
  float* T = t + (size_t)b * HI_ * HI_;
  for (int e = threadIdx.x; e < NTB * NTB; e += 256) {
    int r = e / NTB, c = e % NTB;
    unsigned char m = rm[r][c];
#pragma unroll
    for (int k = 1; k < 9; ++k) m |= rm[r + k][c];
    size_t gi = (size_t)(ty0 + r) * HI_ + (tx0 + c);
    T[gi] = m ? 0.f : S[gi];
  }
}

// out = (mask && inside border) ? out : 0   (in place)
__global__ __launch_bounds__(256) void finalize_kernel(
    float* __restrict__ out, const unsigned char* __restrict__ mask) {
  size_t i = (size_t)blockIdx.x * 256 + threadIdx.x;
  if (i >= (size_t)NPIX) return;
  int Y = (int)((i >> 9) & 511);
  int X = (int)(i & 511);
  bool keep = mask[i] && Y >= 4 && Y < HI_ - 4 && X >= 4 && X < HI_ - 4;
  out[i] = keep ? out[i] : 0.f;
}

// ---------------------------------------------------------------------------
// Host-side orchestration
// ---------------------------------------------------------------------------
extern "C" void kernel_launch(void* const* d_in, const int* in_sizes, int n_in,
                              void* d_out, int out_size, void* d_ws, size_t ws_size,
                              hipStream_t stream) {
  const float* x  = (const float*)d_in[0];   // (32,128,64,64)
  const float* w1 = (const float*)d_in[1];   // (256,128,3,3)
  const float* b1 = (const float*)d_in[2];   // (256,)
  const float* w2 = (const float*)d_in[3];   // (65,256,1,1)
  const float* b2 = (const float*)d_in[4];   // (65,)
  float* scores = (float*)d_out;             // (32,512,512) doubles as NMS scores

  // Workspace layout (all 256B aligned)
  char* ws = (char*)d_ws;
  size_t off = 0;
  __bf16* W1b = (__bf16*)(ws + off); off += (size_t)9 * MID_ * C_ * 2;        // 589824
  __bf16* W2b = (__bf16*)(ws + off); off += (size_t)MPAD_ * MID_ * 2;         // 40960
  float*  b2p = (float*)(ws + off);  off += 512;
  __bf16* Hb  = (__bf16*)(ws + off); off += (size_t)NPOS * MID_ * 2;          // 64 MiB
  float*  t   = (float*)(ws + off);  off += (size_t)NPIX * 4;                 // 32 MiB
  unsigned char* mask = (unsigned char*)(ws + off); off += (size_t)NPIX;      // 8 MiB

  // Weight/bias prep
  prep_w1_kernel<<<(9 * MID_ * C_ + 255) / 256, 256, 0, stream>>>(w1, W1b);
  prep_w2_kernel<<<(MPAD_ * MID_ + 255) / 256, 256, 0, stream>>>(w2, W2b);
  prep_b2_kernel<<<1, 128, 0, stream>>>(b2, b2p);

  // conv1: one block per (b, y) row
  conv1_kernel<<<B_ * H_, 256, 0, stream>>>(x, W1b, b1, Hb);

  // conv2 + softmax + threshold + pixel shuffle -> scores (d_out)
  conv2_kernel<<<NPOS / 128, 256, 0, stream>>>(Hb, W2b, b2p, scores);

  // Grid NMS: mask = local-max(scores); 8 suppress/re-max iterations
  dim3 ngrid(HI_ / NTB, HI_ / NTB, B_);   // (16,16,32)
  nms_max_kernel<<<ngrid, 256, 0, stream>>>(scores, mask, 0);
  for (int it = 0; it < 8; ++it) {
    suppress_kernel<<<ngrid, 256, 0, stream>>>(mask, scores, t);
    nms_max_kernel<<<ngrid, 256, 0, stream>>>(t, mask, 1);
  }

  // kept = mask ? scores : 0, with 4px border zeroed (in place on d_out)
  finalize_kernel<<<(NPIX + 255) / 256, 256, 0, stream>>>(scores, mask);
}
